// MoE_45947560132892
// MI455X (gfx1250) — matrix operand order repaired and checked by reading, not compiled, not run on hardware
//
#include <hip/hip_runtime.h>
#include <hip/hip_bf16.h>
#include <math.h>

#define B_TOK 16384
#define D_DIM 1024
#define H_DIM 256
#define E_EXP 8

// Bbuf geometry: 2 ping-pong tiles of [256 rows][40 bf16] (32 data + 8 pad)
#define BROWS   256
#define BPITCH  40
#define BTILE_E (BROWS * BPITCH)          // elems per tile = 10240
#define BTILE_B (BTILE_E * 2)             // bytes per tile = 20480

typedef __bf16 bf16;
typedef __attribute__((ext_vector_type(16))) __bf16 v16bf;
typedef __attribute__((ext_vector_type(8)))  float  v8f;

union FragU { v16bf v; uint4 q[2]; };

// A-fragment (16x32 bf16, ISA 7.12.2): lane L -> row M=L%16.
// elems 0..7  = K k0+(L<16?0..7:8..15), elems 8..15 = same +16.
__device__ __forceinline__ v16bf load_a_frag(const bf16* base, int ldk, int m, int k0, int lane) {
  const bf16* p = base + m * ldk + k0 + ((lane & 16) ? 8 : 0);
  FragU u;
  u.q[0] = *(const uint4*)(p);
  u.q[1] = *(const uint4*)(p + 16);
  return u.v;
}

// B-fragment (32x16 bf16): lane L -> col N=L%16.
// elems 0..7  = K k0+(L<16?0..7:8..15), elems 8..15 = same +16 (matches A).
__device__ __forceinline__ v16bf load_b_frag(const bf16* base, int ldk, int n, int k0, int lane) {
  const bf16* p = base + n * ldk + k0 + ((lane & 16) ? 8 : 0);
  FragU u;
  u.q[0] = *(const uint4*)(p);
  u.q[1] = *(const uint4*)(p + 16);
  return u.v;
}

// ------------------------------------------------ Tensor Data Mover ---------
#if __has_builtin(__builtin_amdgcn_tensor_load_to_lds)
#define USE_TDM 1
typedef unsigned int u32x4 __attribute__((ext_vector_type(4)));
typedef int          i32x4 __attribute__((ext_vector_type(4)));
typedef int          i32x8 __attribute__((ext_vector_type(8)));

// DMA a [256 rows x 32 cols] bf16 tile (row stride = stride_e elems) into LDS
// at lds_byte_addr, with HW padding of 16B after every 64B so rows land with
// an 80B pitch (== [256][40] layout). ISA 8.3/8.4 descriptor packing.
__device__ __forceinline__ void tdm_load_tile(const bf16* src, unsigned lds_byte_addr,
                                              int stride_e) {
  const unsigned long long ga = (unsigned long long)(uintptr_t)src;
  u32x4 g0;
  g0[0] = 1u;                                                   // count=1 (valid D#)
  g0[1] = lds_byte_addr;                                        // lds_addr
  g0[2] = (unsigned)(ga & 0xFFFFFFFFu);                         // global_addr[31:0]
  g0[3] = (unsigned)((ga >> 32) & 0x01FFFFFFu) | 0x80000000u;   // [56:32] | type=2
  i32x8 g1;
  g1[0] = (1 << 16)      // data_size = 2B
        | (1 << 20)      // pad_enable
        | (3 << 22)      // pad_interval: every 16 DWORDs (64B)
        | (3 << 25);     // pad_amount: 4 DWORDs (16B)
  g1[1] = (stride_e & 0xFFFF) << 16;                            // tensor_dim0[15:0]
  g1[2] = ((stride_e >> 16) & 0xFFFF) | (256 << 16);            // dim0 hi | tensor_dim1=256
  g1[3] = (32 << 16);                                           // tile_dim0 = 32
  g1[4] = 256;                                                  // tile_dim1 = 256 (tile_dim2=0)
  g1[5] = stride_e;                                             // tensor_dim0_stride lo
  g1[6] = 0; g1[7] = 0;                                         // stride hi, dim1_stride = 0
  i32x4 gz = {0, 0, 0, 0};
#if defined(__clang_major__) && __clang_major__ >= 23
  i32x8 gz8 = {0, 0, 0, 0, 0, 0, 0, 0};
  __builtin_amdgcn_tensor_load_to_lds(g0, g1, gz, gz, gz8, 0);
#else
  __builtin_amdgcn_tensor_load_to_lds(g0, g1, gz, gz, 0);
#endif
}
#else
#define USE_TDM 0
#endif

// ---------------------------------------------------------------- init ------
__global__ void init_kernel(float* __restrict__ out, int* __restrict__ counts, int n4) {
  int idx = blockIdx.x * blockDim.x + threadIdx.x;
  if (idx < E_EXP) counts[idx] = 0;
  if (idx < n4) ((float4*)out)[idx] = make_float4(0.f, 0.f, 0.f, 0.f);
}

// --------------------------------------------- weight transpose + bf16 ------
// in: f32 [E][R][C] -> out: bf16 [E][C][R]
__global__ void transpose_convert(const float* __restrict__ in, bf16* __restrict__ out,
                                  int R, int C) {
  __shared__ float tile[32][33];
  const int e = blockIdx.z;
  const float* src = in  + (size_t)e * R * C;
  bf16*        dst = out + (size_t)e * R * C;
  const int c0 = blockIdx.x * 32, r0 = blockIdx.y * 32;
  const int tx = threadIdx.x, ty = threadIdx.y;   // 32 x 8
  #pragma unroll
  for (int i = 0; i < 32; i += 8)
    tile[ty + i][tx] = src[(size_t)(r0 + ty + i) * C + c0 + tx];
  __syncthreads();
  #pragma unroll
  for (int i = 0; i < 32; i += 8)
    dst[(size_t)(c0 + ty + i) * R + r0 + tx] = (bf16)tile[tx][ty + i];
}

// -------------------------------------------------------------- gating ------
// One wave32 per token: logits = x@gate_w + gate_b, top-2, renormalized
// softmax collapses to g1 = 1/(1+exp(l2-l1)). Atomic append to expert lists.
__global__ void gating_kernel(const float* __restrict__ x, const float* __restrict__ gw,
                              const float* __restrict__ gb, int* __restrict__ counts,
                              int* __restrict__ rowidx, float* __restrict__ gates) {
  const int lane = threadIdx.x & 31;
  const int row  = blockIdx.x * 8 + (threadIdx.x >> 5);
  const float* xr = x + (size_t)row * D_DIM;

  float acc[E_EXP];
  #pragma unroll
  for (int j = 0; j < E_EXP; ++j) acc[j] = 0.f;

  for (int it = 0; it < D_DIM / 128; ++it) {
    const int d0 = it * 128 + lane * 4;
    float4 xv = *(const float4*)(xr + d0);
    #pragma unroll
    for (int c = 0; c < 4; ++c) {
      const float xs = (&xv.x)[c];
      float4 w0 = *(const float4*)(gw + (size_t)(d0 + c) * E_EXP);
      float4 w1 = *(const float4*)(gw + (size_t)(d0 + c) * E_EXP + 4);
      acc[0] += xs * w0.x; acc[1] += xs * w0.y; acc[2] += xs * w0.z; acc[3] += xs * w0.w;
      acc[4] += xs * w1.x; acc[5] += xs * w1.y; acc[6] += xs * w1.z; acc[7] += xs * w1.w;
    }
  }
  #pragma unroll
  for (int j = 0; j < E_EXP; ++j) {
    #pragma unroll
    for (int off = 16; off >= 1; off >>= 1) acc[j] += __shfl_xor(acc[j], off, 32);
  }
  if (lane == 0) {
    float l[E_EXP];
    #pragma unroll
    for (int j = 0; j < E_EXP; ++j) l[j] = acc[j] + gb[j];
    int e1 = 0; float m1 = l[0];
    #pragma unroll
    for (int j = 1; j < E_EXP; ++j) if (l[j] > m1) { m1 = l[j]; e1 = j; }
    int e2 = -1; float m2 = -INFINITY;
    #pragma unroll
    for (int j = 0; j < E_EXP; ++j) if (j != e1 && l[j] > m2) { m2 = l[j]; e2 = j; }
    const float g1 = 1.f / (1.f + expf(m2 - m1));
    const float g2 = 1.f - g1;
    int s1 = atomicAdd(&counts[e1], 1);
    rowidx[e1 * B_TOK + s1] = row; gates[e1 * B_TOK + s1] = g1;
    int s2 = atomicAdd(&counts[e2], 1);
    rowidx[e2 * B_TOK + s2] = row; gates[e2 * B_TOK + s2] = g2;
  }
}

// --------------------------------------------- fused expert MLP (WMMA) ------
// Block = 128 threads (4 waves), tile = 64 gathered rows of one expert.
// Weight tiles stream through a double-buffered LDS ping-pong fed by the TDM
// (tensor_load_to_lds, TENSORcnt<=1 keeps one DMA in flight under the WMMAs);
// v_wmma_f32_16x16x32_bf16 compute; gated atomic scatter-add epilogue.
__global__ void __launch_bounds__(128) moe_kernel(
    const float* __restrict__ x, const bf16* __restrict__ W1t, const bf16* __restrict__ W2t,
    const float* __restrict__ b1, const float* __restrict__ b2,
    const int* __restrict__ counts, const int* __restrict__ rowidx,
    const float* __restrict__ gates, float* __restrict__ out) {
  __shared__ __align__(16) bf16 Abuf[64][40];     // gathered x chunk (padded)
  __shared__ __align__(16) bf16 Hbuf[64][264];    // GELU activations (padded)
  __shared__ int   Ridx[64];
  __shared__ float Gval[64];
  extern __shared__ __align__(16) bf16 Bbuf[];    // 2 x [256][40] ping-pong (dynamic LDS)

  const int e    = blockIdx.y;
  const int cnt  = counts[e];
  const int base = blockIdx.x * 64;
  if (base >= cnt) return;
  const int n_rows = min(64, cnt - base);

  const int t    = threadIdx.x;
  const int lane = t & 31;
  const int w    = t >> 5;
  const int l16  = lane & 15;
  const int hi8  = (lane & 16) ? 8 : 0;
#if USE_TDM
  const unsigned ldsB = __builtin_amdgcn_groupstaticsize();  // dynamic-LDS byte offset
#endif

  // stage a weight tile (src = tile origin, row stride = stride_e) into buffer p
  auto stage_b = [&](const bf16* src, int stride_e, int p) {
#if USE_TDM
    if (t < 32) tdm_load_tile(src, ldsB + (unsigned)p * BTILE_B, stride_e);  // wave 0 issues DMA
#else
    bf16* dstb = Bbuf + p * BTILE_E;
    #pragma unroll
    for (int rr = 0; rr < 2; ++rr) {
      const int r = t + rr * 128;
      const uint4* gp = (const uint4*)(src + (size_t)r * stride_e);
      uint4* lp = (uint4*)&dstb[r * BPITCH];
      #pragma unroll
      for (int i = 0; i < 4; ++i) lp[i] = gp[i];
    }
#endif
  };

  if (t < 64) {
    int rv = 0; float gv = 0.f;
    if (t < n_rows) { rv = rowidx[e * B_TOK + base + t]; gv = gates[e * B_TOK + base + t]; }
    Ridx[t] = rv; Gval[t] = gv;
  }
  __syncthreads();

  int myrow[8]; float myg[8]; bool myok[8];
  #pragma unroll
  for (int j = 0; j < 8; ++j) {
    const int rl = w * 16 + j + hi8;      // C/D layout: lanes>=16 hold M=j+8
    myrow[j] = Ridx[rl]; myg[j] = Gval[rl]; myok[j] = (rl < n_rows);
  }

  const bf16* w1e = W1t + (size_t)e * H_DIM * D_DIM;   // [H][D] (= w1^T)
  const bf16* w2e = W2t + (size_t)e * D_DIM * H_DIM;   // [D][H] (= w2^T)

  v8f acc[16];
  #pragma unroll
  for (int nt = 0; nt < 16; ++nt)
    #pragma unroll
    for (int i = 0; i < 8; ++i) acc[nt][i] = 0.f;

  // ---------------- GEMM1 over K = D, 32 per step, pipelined ----------------
  stage_b(w1e, D_DIM, 0);                              // prologue: chunk 0 -> buf 0
  for (int kc = 0; kc < D_DIM / 32; ++kc) {
    const int k0 = kc * 32;
    if (kc + 1 < D_DIM / 32)                           // prefetch next tile
      stage_b(w1e + (kc + 1) * 32, D_DIM, (kc + 1) & 1);
    {   // stage A (overlaps the DMA): 64 rows x 32 cols, f32 -> bf16
      const int ar = t >> 1;
      const int c0 = (t & 1) * 16;
      union { bf16 h[16]; uint4 q[2]; } pk;
      if (ar < n_rows) {
        const float* xp = x + (size_t)Ridx[ar] * D_DIM + k0 + c0;
        #pragma unroll
        for (int i = 0; i < 4; ++i) {
          float4 f = *(const float4*)(xp + i * 4);
          pk.h[i*4+0] = (bf16)f.x; pk.h[i*4+1] = (bf16)f.y;
          pk.h[i*4+2] = (bf16)f.z; pk.h[i*4+3] = (bf16)f.w;
        }
      } else {
        pk.q[0] = make_uint4(0,0,0,0); pk.q[1] = make_uint4(0,0,0,0);
      }
      *(uint4*)&Abuf[ar][c0]     = pk.q[0];
      *(uint4*)&Abuf[ar][c0 + 8] = pk.q[1];
    }
#if USE_TDM
    if (t < 32) {                                      // current tile done; next may fly
      if (kc + 1 < D_DIM / 32) __builtin_amdgcn_s_wait_tensorcnt(1);
      else                     __builtin_amdgcn_s_wait_tensorcnt(0);
    }
#endif
    __syncthreads();
    const bf16* Bp = Bbuf + (kc & 1) * BTILE_E;
    const v16bf afrag = load_a_frag(&Abuf[0][0], 40, w * 16 + l16, 0, lane);
    #pragma unroll
    for (int nt = 0; nt < 16; ++nt) {
      const v16bf bfrag = load_b_frag(Bp, BPITCH, nt * 16 + l16, 0, lane);
      acc[nt] = __builtin_amdgcn_wmma_f32_16x16x32_bf16(
          false, afrag, false, bfrag, (short)0, acc[nt], false, false);
    }
    __syncthreads();
  }

  // first GEMM2 weight tile: DMA hides under the GELU epilogue
  stage_b(w2e, H_DIM, 0);

  // ------------- epilogue 1: bias + exact GELU -> Hbuf (bf16) -------------
  #pragma unroll
  for (int nt = 0; nt < 16; ++nt) {
    const int col = nt * 16 + l16;
    const float bias = b1[e * H_DIM + col];
    #pragma unroll
    for (int j = 0; j < 8; ++j) {
      const float v = acc[nt][j] + bias;
      const float g = 0.5f * v * (1.0f + erff(v * 0.70710678118654752f));
      Hbuf[w * 16 + j + hi8][col] = (bf16)g;
    }
  }

  // ------- GEMM2 over K = H, output D in 4 groups, 32 pipelined steps -------
  for (int st = 0; st < 32; ++st) {
    const int ng = st >> 3;
    const int kc = st & 7;
    const int k0 = kc * 32;
    if (st + 1 < 32) {                                  // prefetch next tile
      const int n2 = (st + 1) >> 3, c2 = (st + 1) & 7;
      stage_b(w2e + (size_t)(n2 * 256) * H_DIM + c2 * 32, H_DIM, (st + 1) & 1);
    }
#if USE_TDM
    if (t < 32) {
      if (st + 1 < 32) __builtin_amdgcn_s_wait_tensorcnt(1);
      else             __builtin_amdgcn_s_wait_tensorcnt(0);
    }
#endif
    __syncthreads();
    if (kc == 0) {
      #pragma unroll
      for (int nt = 0; nt < 16; ++nt)
        #pragma unroll
        for (int i = 0; i < 8; ++i) acc[nt][i] = 0.f;
    }
    const bf16* Bp = Bbuf + (st & 1) * BTILE_E;
    const v16bf afrag = load_a_frag(&Hbuf[0][0], 264, w * 16 + l16, k0, lane);
    #pragma unroll
    for (int nt = 0; nt < 16; ++nt) {
      const v16bf bfrag = load_b_frag(Bp, BPITCH, nt * 16 + l16, 0, lane);
      acc[nt] = __builtin_amdgcn_wmma_f32_16x16x32_bf16(
          false, afrag, false, bfrag, (short)0, acc[nt], false, false);
    }
    __syncthreads();
    if (kc == 7) {
      // ---------- gated scatter-add (2 commutative terms per output elem)
      #pragma unroll
      for (int nt = 0; nt < 16; ++nt) {
        const int col = ng * 256 + nt * 16 + l16;
        const float bias = b2[e * D_DIM + col];
        #pragma unroll
        for (int j = 0; j < 8; ++j) {
          if (myok[j])
            atomicAdd(out + (size_t)myrow[j] * D_DIM + col, myg[j] * (acc[nt][j] + bias));
        }
      }
    }
  }
}

// -------------------------------------------------------------- launch ------
extern "C" void kernel_launch(void* const* d_in, const int* in_sizes, int n_in,
                              void* d_out, int out_size, void* d_ws, size_t ws_size,
                              hipStream_t stream) {
  const float* x   = (const float*)d_in[0];
  const float* gw  = (const float*)d_in[1];
  const float* gb  = (const float*)d_in[2];
  const float* w1  = (const float*)d_in[3];
  const float* b1  = (const float*)d_in[4];
  const float* w2  = (const float*)d_in[5];
  const float* b2  = (const float*)d_in[6];
  float* out = (float*)d_out;

  char* ws = (char*)d_ws;
  bf16*  W1t    = (bf16*)(ws);                                   // 4 MB
  bf16*  W2t    = (bf16*)(ws + (size_t)4 * 1024 * 1024);         // 4 MB
  int*   rowidx = (int*)  (ws + (size_t)8 * 1024 * 1024);        // 512 KB
  float* gates  = (float*)(ws + (size_t)8 * 1024 * 1024 + 512 * 1024); // 512 KB
  int*   counts = (int*)  (ws + (size_t)9 * 1024 * 1024);        // 32 B

  // 1) zero output accumulator + expert counters
  const int n4 = B_TOK * D_DIM / 4;
  init_kernel<<<(n4 + 255) / 256, 256, 0, stream>>>(out, counts, n4);

  // 2) weights -> transposed bf16 (L2-resident, K-contiguous B operands)
  dim3 tb(32, 8);
  transpose_convert<<<dim3(H_DIM / 32, D_DIM / 32, E_EXP), tb, 0, stream>>>(w1, W1t, D_DIM, H_DIM);
  transpose_convert<<<dim3(D_DIM / 32, H_DIM / 32, E_EXP), tb, 0, stream>>>(w2, W2t, H_DIM, D_DIM);

  // 3) top-2 gating + per-expert token compaction
  gating_kernel<<<B_TOK / 8, 256, 0, stream>>>(x, gw, gb, counts, rowidx, gates);

  // 4) fused per-expert MLP: double-buffered TDM weight tiles + bf16 WMMA
  const size_t dynB = 2 * BTILE_B;   // ping-pong Bbuf
  moe_kernel<<<dim3(B_TOK / 64, E_EXP), 128, dynB, stream>>>(
      x, W1t, W2t, b1, b2, counts, rowidx, gates, out);
}
